// MultiHeadAttention_1262720385139
// MI455X (gfx1250) — compile-verified, matrix operand order checked by reading, not run on hardware
//
#include <hip/hip_runtime.h>

// Problem constants (fixed by the reference)
#define SEQ    2048
#define DMODEL 1024
#define NHEAD  16
#define HDIM   64
#define BATCH  2
#define MROWS  (BATCH * SEQ)   // 4096

typedef __attribute__((ext_vector_type(16))) __bf16        v16bf;
typedef __attribute__((ext_vector_type(8)))  float         v8f;
typedef __attribute__((ext_vector_type(4)))  unsigned int  u32x4;

union FragB16 { v16bf v; u32x4 u[2]; };
union Acc     { v8f   v; float f[8]; };

__device__ __forceinline__ void load_frag(FragB16& fr, const __bf16* p) {
  fr.u[0] = *(const u32x4*)(p);
  fr.u[1] = *(const u32x4*)(p + 16);
}

// ---------------------------------------------------------------------------
// fp32 -> bf16 conversion
// ---------------------------------------------------------------------------
__global__ void f32_to_bf16(const float* __restrict__ src,
                            __bf16* __restrict__ dst, int n) {
  int i = blockIdx.x * blockDim.x + threadIdx.x;
  if (i < n) dst[i] = (__bf16)src[i];
}

// ---------------------------------------------------------------------------
// GEMM: out = A (MxK, bf16, row-major) @ W^T (W is NxK bf16 row-major) + bias
// mode 0: bf16 out, layout (B,H,S,HD)
// mode 1: bf16 out, layout (B,H,HD,S)   (transposed V for attention)
// mode 2: f32  out, row-major MxN
// Per wave: 32(M) x 64(N) tile -> 8 accumulators, K stepped by 32.
// ---------------------------------------------------------------------------
__global__ __launch_bounds__(256) void gemm_xwt(
    const __bf16* __restrict__ A,
    const __bf16* __restrict__ W,
    const float*  __restrict__ bias,
    void* __restrict__ outp,
    int mode) {
  const int K = DMODEL, N = DMODEL;
  const int wave   = threadIdx.x >> 5;
  const int lane   = threadIdx.x & 31;
  const int lane16 = lane & 15;
  const int khalf  = (lane >> 4) << 3;   // 0 for lanes 0-15, 8 for 16-31

  const int wid = blockIdx.x * 8 + wave;
  const int tilesN = N / 64;             // 16
  const int m0 = (wid / tilesN) * 32;
  const int n0 = (wid % tilesN) * 64;

  const __bf16* a0p = A + (size_t)(m0 + lane16) * K + khalf;
  const __bf16* a1p = a0p + (size_t)16 * K;
  const __bf16* wp0 = W + (size_t)(n0 + lane16) * K + khalf;

  Acc acc[2][4];
#pragma unroll
  for (int i = 0; i < 2; ++i)
#pragma unroll
    for (int j = 0; j < 4; ++j)
#pragma unroll
      for (int r = 0; r < 8; ++r) acc[i][j].f[r] = 0.0f;

  for (int k0 = 0; k0 < K; k0 += 32) {
    FragB16 a[2], b[4];
    load_frag(a[0], a0p + k0);
    load_frag(a[1], a1p + k0);
#pragma unroll
    for (int j = 0; j < 4; ++j) load_frag(b[j], wp0 + (size_t)j * 16 * K + k0);
#pragma unroll
    for (int i = 0; i < 2; ++i)
#pragma unroll
      for (int j = 0; j < 4; ++j)
        acc[i][j].v = __builtin_amdgcn_wmma_f32_16x16x32_bf16(
            false, a[i].v, false, b[j].v, (short)0, acc[i][j].v, false, false);
  }

#pragma unroll
  for (int i = 0; i < 2; ++i) {
#pragma unroll
    for (int j = 0; j < 4; ++j) {
      const int n  = n0 + j * 16 + lane16;
      const float bb = bias[n];
#pragma unroll
      for (int r = 0; r < 8; ++r) {
        const int m = m0 + i * 16 + r + khalf;   // C layout: +8 rows for hi lanes
        const float val = acc[i][j].f[r] + bb;
        if (mode == 2) {
          ((float*)outp)[(size_t)m * N + n] = val;
        } else {
          const int b_  = m >> 11, s  = m & (SEQ - 1);
          const int h   = n >> 6,  hd = n & (HDIM - 1);
          size_t idx;
          if (mode == 0)
            idx = (((size_t)(b_ * NHEAD + h) * SEQ + s) * HDIM + hd);
          else
            idx = (((size_t)(b_ * NHEAD + h) * HDIM + hd) * SEQ + s);
          ((__bf16*)outp)[idx] = (__bf16)val;
        }
      }
    }
  }
}

// ---------------------------------------------------------------------------
// Flash attention: one wave handles a 16-row Q tile for one (b,h).
// Q,K: (B*H, S, HD) bf16.  Vt: (B*H, HD, S) bf16.  Ctx out: (B, S, D) bf16.
// Online softmax over 32-key blocks; causal + attention_mask.
// ---------------------------------------------------------------------------
__global__ __launch_bounds__(256) void flash_attn(
    const __bf16* __restrict__ Q,
    const __bf16* __restrict__ Kmat,
    const __bf16* __restrict__ Vt,
    const int*    __restrict__ am,
    __bf16* __restrict__ Ctx) {
  __shared__ alignas(16) __bf16 plds[8][16 * 32];

  const int wave   = threadIdx.x >> 5;
  const int lane   = threadIdx.x & 31;
  const int lane16 = lane & 15;
  const int khalf  = (lane >> 4) << 3;

  const int wid = blockIdx.x * 8 + wave;
  const int qt  = wid & (SEQ / 16 - 1);   // 0..127
  const int bh  = wid >> 7;               // 0..31
  const int b_  = bh >> 4;
  const int h   = bh & 15;
  const int q0  = qt * 16;

  const __bf16* Qb = Q    + (size_t)bh * SEQ * HDIM;
  const __bf16* Kb = Kmat + (size_t)bh * SEQ * HDIM;
  const __bf16* Vb = Vt   + (size_t)bh * HDIM * SEQ;

  FragB16 qa[2];
  {
    const __bf16* qp = Qb + (size_t)(q0 + lane16) * HDIM + khalf;
    load_frag(qa[0], qp);
    load_frag(qa[1], qp + 32);
  }

  float mrow[8], lrow[8];
  int amq[8];
#pragma unroll
  for (int r = 0; r < 8; ++r) {
    mrow[r] = -3.0e38f;
    lrow[r] = 0.0f;
    amq[r]  = am[b_ * SEQ + q0 + r + khalf];
  }
  Acc o[4];
#pragma unroll
  for (int c = 0; c < 4; ++c)
#pragma unroll
    for (int r = 0; r < 8; ++r) o[c].f[r] = 0.0f;

  const float scale = 0.125f;  // 1/sqrt(64)

  for (int kb = 0; kb < q0 + 16; kb += 32) {
    // ---- S = Q @ K^T for 32 keys (two 16-wide N subtiles) ----
    Acc s[2];
#pragma unroll
    for (int sub = 0; sub < 2; ++sub)
#pragma unroll
      for (int r = 0; r < 8; ++r) s[sub].f[r] = 0.0f;

#pragma unroll
    for (int sub = 0; sub < 2; ++sub) {
      FragB16 kf0, kf1;
      const __bf16* kp = Kb + (size_t)(kb + sub * 16 + lane16) * HDIM + khalf;
      load_frag(kf0, kp);
      load_frag(kf1, kp + 32);
      s[sub].v = __builtin_amdgcn_wmma_f32_16x16x32_bf16(
          false, qa[0].v, false, kf0.v, (short)0, s[sub].v, false, false);
      s[sub].v = __builtin_amdgcn_wmma_f32_16x16x32_bf16(
          false, qa[1].v, false, kf1.v, (short)0, s[sub].v, false, false);
    }

    const int amk0 = am[b_ * SEQ + kb + lane16];
    const int amk1 = am[b_ * SEQ + kb + 16 + lane16];

    // ---- scale + mask + online softmax ----
#pragma unroll
    for (int r = 0; r < 8; ++r) {
      const int qrow = q0 + r + khalf;
      {
        const int kc0 = kb + lane16;
        const int kc1 = kb + 16 + lane16;
        const bool ok0 = (kc0 <= qrow) && amk0 && amq[r];
        const bool ok1 = (kc1 <= qrow) && amk1 && amq[r];
        s[0].f[r] = ok0 ? s[0].f[r] * scale : -3.0e38f;
        s[1].f[r] = ok1 ? s[1].f[r] * scale : -3.0e38f;
      }
      float t = fmaxf(s[0].f[r], s[1].f[r]);
      t = fmaxf(t, __shfl_xor(t, 1));
      t = fmaxf(t, __shfl_xor(t, 2));
      t = fmaxf(t, __shfl_xor(t, 4));
      t = fmaxf(t, __shfl_xor(t, 8));     // row max within 16-lane half group
      const float mnew  = fmaxf(mrow[r], t);
      const float alpha = __expf(mrow[r] - mnew);
      const float p0 = __expf(s[0].f[r] - mnew);
      const float p1 = __expf(s[1].f[r] - mnew);
      s[0].f[r] = p0;
      s[1].f[r] = p1;
      float rs = p0 + p1;
      rs += __shfl_xor(rs, 1);
      rs += __shfl_xor(rs, 2);
      rs += __shfl_xor(rs, 4);
      rs += __shfl_xor(rs, 8);            // row sum
      lrow[r] = lrow[r] * alpha + rs;
      mrow[r] = mnew;
#pragma unroll
      for (int c = 0; c < 4; ++c) o[c].f[r] *= alpha;
    }

    // ---- C-layout P -> A-fragment via per-wave LDS staging ----
#pragma unroll
    for (int r = 0; r < 8; ++r) {
      const int row = r + khalf;
      plds[wave][row * 32 + lane16]      = (__bf16)s[0].f[r];
      plds[wave][row * 32 + 16 + lane16] = (__bf16)s[1].f[r];
    }
    FragB16 pa;
    load_frag(pa, &plds[wave][lane16 * 32 + khalf]);

    // ---- O += P @ V (V transposed: B columns are contiguous in S) ----
#pragma unroll
    for (int c = 0; c < 4; ++c) {
      FragB16 vf;
      load_frag(vf, Vb + (size_t)(c * 16 + lane16) * SEQ + kb + khalf);
      o[c].v = __builtin_amdgcn_wmma_f32_16x16x32_bf16(
          false, pa.v, false, vf.v, (short)0, o[c].v, false, false);
    }
  }

  // ---- normalize and write ctx (B,S,D) bf16 ----
#pragma unroll
  for (int r = 0; r < 8; ++r) {
    const float inv  = 1.0f / lrow[r];
    const int   srow = q0 + r + khalf;
    const size_t base = ((size_t)(b_ * SEQ + srow)) * DMODEL + (size_t)h * HDIM;
#pragma unroll
    for (int c = 0; c < 4; ++c)
      Ctx[base + c * 16 + lane16] = (__bf16)(o[c].f[r] * inv);
  }
}

// ---------------------------------------------------------------------------
// Launch pipeline
// ---------------------------------------------------------------------------
extern "C" void kernel_launch(void* const* d_in, const int* in_sizes, int n_in,
                              void* d_out, int out_size, void* d_ws, size_t ws_size,
                              hipStream_t stream) {
  (void)in_sizes; (void)n_in; (void)out_size; (void)ws_size;

  const float* x  = (const float*)d_in[0];
  const int*   am = (const int*)  d_in[1];
  const float* Wq = (const float*)d_in[2];
  const float* bq = (const float*)d_in[3];
  const float* Wk = (const float*)d_in[4];
  const float* bk = (const float*)d_in[5];
  const float* Wv = (const float*)d_in[6];
  const float* bv = (const float*)d_in[7];
  const float* Wp = (const float*)d_in[8];
  const float* bp = (const float*)d_in[9];

  char* ws = (char*)d_ws;
  const size_t MB = 1024 * 1024;
  __bf16* xbf  = (__bf16*)(ws +  0 * MB);  // 8 MB
  __bf16* wqb  = (__bf16*)(ws +  8 * MB);  // 2 MB
  __bf16* wkb  = (__bf16*)(ws + 10 * MB);
  __bf16* wvb  = (__bf16*)(ws + 12 * MB);
  __bf16* wpb  = (__bf16*)(ws + 14 * MB);
  __bf16* qb   = (__bf16*)(ws + 16 * MB);  // 8 MB  (B,H,S,HD)
  __bf16* kbuf = (__bf16*)(ws + 24 * MB);  // 8 MB  (B,H,S,HD)
  __bf16* vtb  = (__bf16*)(ws + 32 * MB);  // 8 MB  (B,H,HD,S)
  __bf16* ctx  = (__bf16*)(ws + 40 * MB);  // 8 MB  (B,S,D)

  const int nx = MROWS * DMODEL;    // 4,194,304
  const int nw = DMODEL * DMODEL;   // 1,048,576

  f32_to_bf16<<<(nx + 255) / 256, 256, 0, stream>>>(x,  xbf, nx);
  f32_to_bf16<<<(nw + 255) / 256, 256, 0, stream>>>(Wq, wqb, nw);
  f32_to_bf16<<<(nw + 255) / 256, 256, 0, stream>>>(Wk, wkb, nw);
  f32_to_bf16<<<(nw + 255) / 256, 256, 0, stream>>>(Wv, wvb, nw);
  f32_to_bf16<<<(nw + 255) / 256, 256, 0, stream>>>(Wp, wpb, nw);

  // QKV projections: 2048 wave tiles (128 M-tiles x 16 N-tiles) / 8 waves
  gemm_xwt<<<256, 256, 0, stream>>>(xbf, wqb, bq, (void*)qb,   0);
  gemm_xwt<<<256, 256, 0, stream>>>(xbf, wkb, bk, (void*)kbuf, 0);
  gemm_xwt<<<256, 256, 0, stream>>>(xbf, wvb, bv, (void*)vtb,  1);

  // attention: B*H*(S/16) = 4096 wave tiles / 8 waves per block
  flash_attn<<<512, 256, 0, stream>>>(qb, kbuf, vtb, am, ctx);

  // output projection -> fp32 d_out
  gemm_xwt<<<256, 256, 0, stream>>>(ctx, wpb, bp, d_out, 2);
}
